// PNAConv_86277303042055
// MI455X (gfx1250) — matrix operand order repaired and checked by reading, not compiled, not run on hardware
//
#include <hip/hip_runtime.h>

typedef __attribute__((ext_vector_type(16))) _Float16 v16h;
typedef __attribute__((ext_vector_type(8)))  float    v8f;

#define N_NODES 50000
#define N_EDGES 800000
#define C_IN    64
#define TOWERS  4
#define F_IN    16
#define OUT_CH  128
#define AVG_LOG_DEG 2.8332133441f /* log(17) */

// Interleaved per-channel stats: stat[(node*64 + ch)*4 + {0:sum, 1:sumsq, 2:max, 3:min}].
// One base address serves all four atomic updates via the 24-bit immediate offset.
// Native CDNA5 float max/min atomics (IEEE maximumNumber/minimumNumber, ISA 15.18.3
// opcodes 81/82); the toolchain only emits CAS loops from HIP atomics, so use asm.
static __device__ __forceinline__ void atomic_maxmin_f32(float* base, float v) {
    unsigned long long a = (unsigned long long)(uintptr_t)base;
    asm volatile("global_atomic_max_num_f32 %0, %1, off offset:8 scope:SCOPE_DEV\n\t"
                 "global_atomic_min_num_f32 %0, %1, off offset:12 scope:SCOPE_DEV"
                 :: "v"(a), "v"(v) : "memory");
}

// ---------------- init: {0, 0, -inf, +inf} per channel, zero counts ----------------
__global__ __launch_bounds__(256) void init_ws_k(float4* __restrict__ stat4, float* __restrict__ cnt) {
    int i = blockIdx.x * 256 + threadIdx.x;
    if (i < N_NODES * C_IN)
        stat4[i] = make_float4(0.f, 0.f, -__builtin_inff(), __builtin_inff());
    if (i < N_NODES) cnt[i] = 0.f;
}

// ---------------- per-edge receiver degree ----------------
__global__ __launch_bounds__(256) void count_k(const int* __restrict__ rcv, float* __restrict__ cnt) {
    int i = blockIdx.x * 256 + threadIdx.x;
    if (i < N_EDGES) atomicAdd(&cnt[rcv[i]], 1.0f);
}

// ---------------- edge kernel: gather + pre-MLP (WMMA) + atomic multi-aggregation ----------------
// One wave per 16-edge tile; per tower: A = [16 edges x 32(xi||xj)] f16, B = pre_w[t] 32x16 f16.
__global__ __launch_bounds__(256) void edge_k(const float* __restrict__ nodes,
                                              const int* __restrict__ snd, const int* __restrict__ rcv,
                                              const float* __restrict__ pre_w, const float* __restrict__ pre_b,
                                              float* __restrict__ stat) {
    __shared__ _Float16 Wsh[TOWERS * 32 * 16];  // [t][k][o], 4 KB
    for (int idx = threadIdx.x; idx < TOWERS * 32 * 16; idx += 256)
        Wsh[idx] = (_Float16)pre_w[idx];
    __syncthreads();

    const int wave = threadIdx.x >> 5;
    const int lane = threadIdx.x & 31;
    const int m  = lane & 15;   // A row / B,D column
    const int hi = lane >> 4;
    const int tileBase = (blockIdx.x * 8 + wave) * 16;  // E % 128 == 0

    const int e0 = tileBase + m;
    const int s = snd[e0];
    const int r = rcv[e0];

    int r2[8];
#pragma unroll
    for (int v = 0; v < 8; ++v) r2[v] = rcv[tileBase + v + hi * 8];

#pragma unroll
    for (int t = 0; t < TOWERS; ++t) {
        // A fragment: lane needs h(m, k) for k = hi*8..hi*8+7 (x_i) and 16+hi*8.. (x_j)
        const float* xs = nodes + (long)s * C_IN + t * F_IN + hi * 8;
        const float* xr = nodes + (long)r * C_IN + t * F_IN + hi * 8;
        v16h a, b;
#pragma unroll
        for (int q = 0; q < 8; ++q) {
            a[q]     = (_Float16)xs[q];
            a[8 + q] = (_Float16)xr[q];
        }
        // B fragment: 32x16, lane col = m, k = (v>>2)*16 + hi*8 + 2(v&3) (+1)
#pragma unroll
        for (int v = 0; v < 8; ++v) {
            int kb = ((v >> 2) << 4) + hi * 8 + ((v & 3) << 1);
            b[2 * v]     = Wsh[(t * 32 + kb) * 16 + m];
            b[2 * v + 1] = Wsh[(t * 32 + kb + 1) * 16 + m];
        }
        float bias = pre_b[t * F_IN + m];
        v8f c;
#pragma unroll
        for (int v = 0; v < 8; ++v) c[v] = bias;

        c = __builtin_amdgcn_wmma_f32_16x16x32_f16(false, a, false, b, (short)0, c, false, false);

        const int col = t * F_IN + m;
#pragma unroll
        for (int v = 0; v < 8; ++v) {
            float val = c[v];                 // msg for edge (tileBase + v + 8*hi), channel col
            float* base = stat + ((long)r2[v] * C_IN + col) * 4;
            atomicAdd(base + 0, val);         // folds to offset:0
            atomicAdd(base + 1, val * val);   // folds to offset:4
            atomic_maxmin_f32(base, val);     // offset:8 / offset:12, shared address reg
        }
    }
}

// ---------------- node kernel: build PNA features in LDS, post-MLP via 7x2 WMMAs ----------------
// One wave per (16-node tile, tower). Output mid[N,128] stored as f16.
__global__ __launch_bounds__(32) void node_k(const float* __restrict__ nodes,
                                             const float* __restrict__ post_w, const float* __restrict__ post_b,
                                             const float4* __restrict__ stat4,
                                             const float* __restrict__ cnt, _Float16* __restrict__ mid) {
    __shared__ _Float16 Wsh[224 * 32];     // post_w[t] padded K 208->224, 14 KB
    __shared__ _Float16 feat[16][224];     // 7 KB

    const int t    = blockIdx.x & 3;
    const int tile = blockIdx.x >> 2;      // 50000/16 = 3125 exact
    const int lane = threadIdx.x;

    for (int idx = lane; idx < 224 * 32; idx += 32) {
        int k = idx >> 5, o = idx & 31;
        Wsh[idx] = (k < 208) ? (_Float16)post_w[((long)t * 208 + k) * 32 + o] : (_Float16)0.f;
    }

    // Build feature rows: 2 lanes per node, 8 channels each.
    const int mrow = lane >> 1;
    const int seg  = lane & 1;
    const long n   = (long)tile * 16 + mrow;
    const float c0   = cnt[n];
    const float d    = fmaxf(c0, 1.0f);
    const float invd = 1.0f / d;
    const float logd = logf(d + 1.0f);
    const float amp  = logd / AVG_LOG_DEG;
    const float att  = AVG_LOG_DEG / logd;
    const bool  has  = c0 > 0.f;

#pragma unroll
    for (int i = 0; i < 8; ++i) {
        const int  c   = seg * 8 + i;
        const long off = n * C_IN + t * F_IN + c;   // same offset for x and per-channel stats
        const float4 q = stat4[off];                // single b128 load: sum, sumsq, max, min
        float mean = q.x * invd;
        float msq  = q.y * invd;
        float stdv = sqrtf(fmaxf(msq - mean * mean, 0.f) + 1e-5f);
        float mx = has ? q.z : 0.f;
        float mn = has ? q.w : 0.f;
        feat[mrow][c]        = (_Float16)nodes[off];
        feat[mrow][16 + c]   = (_Float16)mean;
        feat[mrow][32 + c]   = (_Float16)stdv;
        feat[mrow][48 + c]   = (_Float16)mx;
        feat[mrow][64 + c]   = (_Float16)mn;
        feat[mrow][80 + c]   = (_Float16)(mean * amp);
        feat[mrow][96 + c]   = (_Float16)(stdv * amp);
        feat[mrow][112 + c]  = (_Float16)(mx * amp);
        feat[mrow][128 + c]  = (_Float16)(mn * amp);
        feat[mrow][144 + c]  = (_Float16)(mean * att);
        feat[mrow][160 + c]  = (_Float16)(stdv * att);
        feat[mrow][176 + c]  = (_Float16)(mx * att);
        feat[mrow][192 + c]  = (_Float16)(mn * att);
        feat[mrow][208 + c]  = (_Float16)0.f;       // K padding
    }
    __syncthreads();

    const int m  = lane & 15;
    const int hi = lane >> 4;
#pragma unroll
    for (int ch = 0; ch < 2; ++ch) {               // two 16-wide output column halves
        float bias = post_b[t * 32 + ch * 16 + m];
        v8f acc;
#pragma unroll
        for (int v = 0; v < 8; ++v) acc[v] = bias;
#pragma unroll
        for (int j = 0; j < 7; ++j) {              // K = 224 in steps of 32
            v16h a, b;
#pragma unroll
            for (int v = 0; v < 8; ++v) {
                int ka = 32 * j + hi * 16 + v * 2;
                a[2 * v]     = feat[m][ka];
                a[2 * v + 1] = feat[m][ka + 1];
                int kb = 32 * j + hi * 16 + v * 2;
                b[2 * v]     = Wsh[kb * 32 + ch * 16 + m];
                b[2 * v + 1] = Wsh[(kb + 1) * 32 + ch * 16 + m];
            }
            acc = __builtin_amdgcn_wmma_f32_16x16x32_f16(false, a, false, b, (short)0, acc, false, false);
        }
#pragma unroll
        for (int v = 0; v < 8; ++v) {
            long node = (long)tile * 16 + v + hi * 8;
            mid[node * OUT_CH + t * 32 + ch * 16 + m] = (_Float16)acc[v];
        }
    }
}

// ---------------- final kernel: [N,128] @ lin_w[128,128] + lin_b, 4 WMMAs per wave ----------------
__global__ __launch_bounds__(32) void final_k(const _Float16* __restrict__ mid,
                                              const float* __restrict__ lin_w, const float* __restrict__ lin_b,
                                              float* __restrict__ out) {
    __shared__ _Float16 Lsh[128 * 16];  // lin_w column slice, 4 KB

    const int g    = blockIdx.x & 7;    // 16-wide output column group
    const int tile = blockIdx.x >> 3;
    const int lane = threadIdx.x;

    for (int idx = lane; idx < 128 * 16; idx += 32) {
        int k = idx >> 4, j = idx & 15;
        Lsh[idx] = (_Float16)lin_w[(long)k * OUT_CH + g * 16 + j];
    }
    __syncthreads();

    const int m  = lane & 15;
    const int hi = lane >> 4;
    float bias = lin_b[g * 16 + m];
    v8f acc;
#pragma unroll
    for (int v = 0; v < 8; ++v) acc[v] = bias;

#pragma unroll
    for (int j = 0; j < 4; ++j) {       // K = 128
        v16h a, b;
        const long row = (long)tile * 16 + m;
#pragma unroll
        for (int v = 0; v < 8; ++v) {
            int ka = 32 * j + hi * 16 + v * 2;
            a[2 * v]     = mid[row * OUT_CH + ka];
            a[2 * v + 1] = mid[row * OUT_CH + ka + 1];
            int kb = 32 * j + hi * 16 + v * 2;
            b[2 * v]     = Lsh[kb * 16 + m];
            b[2 * v + 1] = Lsh[(kb + 1) * 16 + m];
        }
        acc = __builtin_amdgcn_wmma_f32_16x16x32_f16(false, a, false, b, (short)0, acc, false, false);
    }
#pragma unroll
    for (int v = 0; v < 8; ++v) {
        long node = (long)tile * 16 + v + hi * 8;
        out[node * OUT_CH + g * 16 + m] = acc[v];
    }
}

extern "C" void kernel_launch(void* const* d_in, const int* in_sizes, int n_in,
                              void* d_out, int out_size, void* d_ws, size_t ws_size,
                              hipStream_t stream) {
    const float* nodes  = (const float*)d_in[0];
    const int*   snd    = (const int*)d_in[1];
    const int*   rcv    = (const int*)d_in[2];
    const float* pre_w  = (const float*)d_in[3];
    const float* pre_b  = (const float*)d_in[4];
    const float* post_w = (const float*)d_in[5];
    const float* post_b = (const float*)d_in[6];
    const float* lin_w  = (const float*)d_in[7];
    const float* lin_b  = (const float*)d_in[8];

    char* ws = (char*)d_ws;
    const size_t statN = (size_t)N_NODES * C_IN;
    float* stat = (float*)ws;   ws += statN * 4 * sizeof(float);     // interleaved sum/sumsq/max/min
    float* cnt  = (float*)ws;   ws += (((size_t)N_NODES * sizeof(float)) + 255) & ~(size_t)255;
    _Float16* mid = (_Float16*)ws; // N*128 halves

    const int M = N_NODES * C_IN;
    hipLaunchKernelGGL(init_ws_k, dim3((M + 255) / 256), dim3(256), 0, stream, (float4*)stat, cnt);
    hipLaunchKernelGGL(count_k, dim3((N_EDGES + 255) / 256), dim3(256), 0, stream, rcv, cnt);
    hipLaunchKernelGGL(edge_k, dim3(N_EDGES / 128), dim3(256), 0, stream,
                       nodes, snd, rcv, pre_w, pre_b, stat);
    hipLaunchKernelGGL(node_k, dim3((N_NODES / 16) * TOWERS), dim3(32), 0, stream,
                       nodes, post_w, post_b, (const float4*)stat, cnt, mid);
    hipLaunchKernelGGL(final_k, dim3((N_NODES / 16) * 8), dim3(32), 0, stream,
                       mid, lin_w, lin_b, (float*)d_out);
}
